// WSDNetDenseAdasoft_48885317763603
// MI455X (gfx1250) — matrix operand order repaired and checked by reading, not compiled, not run on hardware
//
#include <hip/hip_runtime.h>
#include <hip/hip_bf16.h>

typedef __attribute__((ext_vector_type(16))) _Float16 v16h;
typedef __attribute__((ext_vector_type(8)))  _Float16 v8h;
typedef __attribute__((ext_vector_type(8)))  float    v8f;

#define N_ROWS   512
#define D_IN     1024
#define HID      512
#define C0_      1000
#define HEADN    1003
#define S_TOT    117660
#define T3_WSD   107660
#define T3_SLM   170000
#define NEG_BIG  (-3.4e38f)

// ---------------- BatchNorm ----------------
__global__ void bn_stats_kernel(const float* __restrict__ x, float* __restrict__ mu,
                                float* __restrict__ rsig) {
    int d = blockIdx.x * blockDim.x + threadIdx.x;
    if (d >= D_IN) return;
    float s = 0.f, ss = 0.f;
    for (int n = 0; n < N_ROWS; ++n) {
        float v = x[n * D_IN + d];
        s += v; ss += v * v;
    }
    float m = s * (1.0f / N_ROWS);
    float var = ss * (1.0f / N_ROWS) - m * m;
    mu[d] = m;
    rsig[d] = rsqrtf(var + 1e-5f);
}

__global__ void bn_apply_kernel(const float* __restrict__ x, const float* __restrict__ mu,
                                const float* __restrict__ rsig, const float* __restrict__ g,
                                const float* __restrict__ b, _Float16* __restrict__ xh) {
    int i = blockIdx.x * blockDim.x + threadIdx.x;
    if (i >= N_ROWS * D_IN) return;
    int d = i & (D_IN - 1);
    xh[i] = (_Float16)((x[i] - mu[d]) * rsig[d] * g[d] + b[d]);
}

// ---------------- f32 -> f16 convert ----------------
__global__ void f2h_kernel(const float* __restrict__ src, _Float16* __restrict__ dst, int n) {
    int i = blockIdx.x * blockDim.x + threadIdx.x;
    if (i < n) dst[i] = (_Float16)src[i];
}

// ---------------- WMMA GEMM: C[M,N] = A[M,K] @ W[N,K]^T (+bias)(+relu) ----------------
// one wave per 16x16 output tile; K multiple of 32; M multiple of 16; N arbitrary.
// Out-of-range B columns are CLAMPED (not zeroed): WMMA columns are independent and
// clamped columns are discarded at the guarded store, so the inner loop stays
// divergence-free (no exec-mask churn around the B loads).
__global__ __launch_bounds__(256) void wmma_gemm_kernel(
    const _Float16* __restrict__ A, const _Float16* __restrict__ W,
    const float* __restrict__ bias, float* __restrict__ C, _Float16* __restrict__ Ch,
    int M, int N, int K, int relu) {
    int wave = blockIdx.x * (blockDim.x >> 5) + (threadIdx.x >> 5);
    int lane = threadIdx.x & 31;
    int tilesN = (N + 15) >> 4;
    int tiles = (M >> 4) * tilesN;
    if (wave >= tiles) return;                 // wave-uniform

    int tm = wave / tilesN, tn = wave % tilesN;
    int row0 = tm << 4, col0 = tn << 4;
    int r    = lane & 15;
    int half = lane >> 4;                      // 0: lanes 0-15, 1: lanes 16-31

    int wrow = col0 + r;
    int wrowc = wrow < N ? wrow : (N - 1);     // clamp: garbage cols never stored

    const _Float16* ap = A + (size_t)(row0 + r) * K + half * 16;
    const _Float16* bp = W + (size_t)wrowc * K + half * 16;

    v8f acc;
#pragma unroll
    for (int i = 0; i < 8; ++i) acc[i] = 0.f;

    for (int k0 = 0; k0 < K; k0 += 32) {
        // A fragment (16x32 f16): lane half 0 -> K 0..15 ; half 1 -> K 16..31 (contiguous)
        v8h lo = *(const v8h*)(ap + k0);
        v8h hi = *(const v8h*)(ap + k0 + 8);
        v16h a;
#pragma unroll
        for (int j = 0; j < 8; ++j) { a[j] = lo[j]; a[8 + j] = hi[j]; }

        // B fragment (32x16 f16): lane half 0 -> K 0..15 ; half 1 -> K 16..31 (contiguous)
        v16h b = *(const v16h*)(bp + k0);

        acc = __builtin_amdgcn_wmma_f32_16x16x32_f16(false, a, false, b,
                                                     (short)0, acc, false, false);
    }

    // C/D layout: lane half h -> rows row0 + 8h + i (i=0..7), col = col0 + (lane&15)
    int ccol  = col0 + r;
    int crow0 = row0 + half * 8;
    if (ccol < N) {
        float bv = bias ? bias[ccol] : 0.f;
#pragma unroll
        for (int i = 0; i < 8; ++i) {
            float v = acc[i] + bv;
            if (relu) v = fmaxf(v, 0.f);
            size_t idx = (size_t)(crow0 + i) * N + ccol;
            if (C)  C[idx]  = v;
            if (Ch) Ch[idx] = (_Float16)v;
        }
    }
}

// ---------------- online log-sum-exp reduction helpers ----------------
__device__ __forceinline__ void lse_update(float v, float& mx, float& sm) {
    if (v > mx) { sm = sm * __expf(mx - v) + 1.f; mx = v; }
    else        { sm += __expf(v - mx); }
}

__device__ __forceinline__ float block_lse(float mx, float sm, float* smx, float* ssm) {
    int t = threadIdx.x;
    smx[t] = mx; ssm[t] = sm;
    __syncthreads();
    for (int off = 128; off > 0; off >>= 1) {
        if (t < off) {
            float m1 = smx[t], s1 = ssm[t];
            float m2 = smx[t + off], s2 = ssm[t + off];
            float m = fmaxf(m1, m2);
            ssm[t] = s1 * __expf(m1 - m) + s2 * __expf(m2 - m);
            smx[t] = m;
        }
        __syncthreads();
    }
    return smx[0] + logf(ssm[0]);
}

// head LSE over [N_ROWS, HEADN]
__global__ __launch_bounds__(256) void row_lse_kernel(const float* __restrict__ logits,
                                                      float* __restrict__ lse) {
    __shared__ float smx[256], ssm[256];
    int n = blockIdx.x, t = threadIdx.x;
    const float* r = logits + (size_t)n * HEADN;
    float mx = NEG_BIG, sm = 0.f;
    for (int j = t; j < HEADN; j += 256) lse_update(r[j], mx, sm);
    float v = block_lse(mx, sm, smx, ssm);
    if (t == 0) lse[n] = v;
}

// tail cluster LSE via on-the-fly dots; writes tadd[n*3+tier] = headlog[n,C0+tier] - head_lse[n] - lse
__global__ __launch_bounds__(256) void tail_lse_kernel(
    const float* __restrict__ hp, int L, const float* __restrict__ w2, int J,
    const float* __restrict__ headlog, const float* __restrict__ head_lse,
    int tier, float* __restrict__ tadd) {
    __shared__ float smx[256], ssm[256];
    __shared__ float hps[128];
    int n = blockIdx.x, t = threadIdx.x;
    for (int i = t; i < L; i += 256) hps[i] = hp[(size_t)n * L + i];
    __syncthreads();
    float mx = NEG_BIG, sm = 0.f;
    for (int j = t; j < J; j += 256) {
        const float* wr = w2 + (size_t)j * L;
        float d = 0.f;
        for (int l = 0; l < L; ++l) d += hps[l] * wr[l];
        lse_update(d, mx, sm);
    }
    float v = block_lse(mx, sm, smx, ssm);
    if (t == 0)
        tadd[n * 3 + tier] = headlog[(size_t)n * HEADN + C0_ + tier] - head_lse[n] - v;
}

// ---------------- fused output kernel ----------------
struct FinalArgs {
    const float *wsd_headlog, *wsd_lseh, *wsd_tadd;
    const float *wsd_hp1, *wsd_hp2, *wsd_hp3;
    const float *wsd_w21, *wsd_w22, *wsd_w23;
    const float *slm_headlog, *slm_lseh, *slm_tadd;
    const float *slm_hp1, *slm_hp2, *slm_hp3;
    const float *slm_w21, *slm_w22, *slm_w23;
    const float *sv_vals;
    const int   *sv_cols;
    float       *out;
};

__device__ __forceinline__ float dotL(const float* __restrict__ a,
                                      const float* __restrict__ b, int L) {
    float s = 0.f;
    for (int l = 0; l < L; ++l) s += a[l] * b[l];
    return s;
}

__device__ __forceinline__ float dot8(const float* __restrict__ a,
                                      const float* __restrict__ b) {
    float s = 0.f;
#pragma unroll
    for (int l = 0; l < 8; ++l) s += a[l] * b[l];
    return s;
}

__global__ __launch_bounds__(256) void final_out_kernel(FinalArgs A) {
    __shared__ float sh_whp1[128], sh_whp2[32], sh_whp3[8];
    __shared__ float sh_lhp1[128], sh_lhp2[32], sh_lhp3[8];
    __shared__ float sh_slm_head[C0_];   // already lse-subtracted

    int n = blockIdx.y;
    int t = threadIdx.x;
    int s = blockIdx.x * 256 + t;

    if (t < 128) { sh_whp1[t] = A.wsd_hp1[n * 128 + t]; sh_lhp1[t] = A.slm_hp1[n * 128 + t]; }
    if (t < 32)  { sh_whp2[t] = A.wsd_hp2[n * 32 + t];  sh_lhp2[t] = A.slm_hp2[n * 32 + t]; }
    if (t < 8)   { sh_whp3[t] = A.wsd_hp3[n * 8 + t];   sh_lhp3[t] = A.slm_hp3[n * 8 + t]; }
    float slm_lse = A.slm_lseh[n];
    for (int j = t; j < C0_; j += 256)
        sh_slm_head[j] = A.slm_headlog[(size_t)n * HEADN + j] - slm_lse;
    __syncthreads();

    if (s >= S_TOT) return;

    // WSD adaptive log-prob y[n,s]
    float y;
    if (s < 1000)       y = A.wsd_headlog[(size_t)n * HEADN + s] - A.wsd_lseh[n];
    else if (s < 3000)  y = dotL(sh_whp1, A.wsd_w21 + (size_t)(s - 1000) * 128, 128) + A.wsd_tadd[n * 3 + 0];
    else if (s < 10000) y = dotL(sh_whp2, A.wsd_w22 + (size_t)(s - 3000) * 32, 32)   + A.wsd_tadd[n * 3 + 1];
    else                y = dot8(sh_whp3, A.wsd_w23 + (size_t)(s - 10000) * 8)        + A.wsd_tadd[n * 3 + 2];

    // sparse mixture of SLM log-probs (values reconstructed on the fly)
    float acc = 0.f;
#pragma unroll
    for (int k = 0; k < 8; ++k) {
        int   col = A.sv_cols[(size_t)s * 8 + k];
        float val = A.sv_vals[(size_t)s * 8 + k];
        float v;
        if (col < 1000)       v = sh_slm_head[col];
        else if (col < 3000)  v = dotL(sh_lhp1, A.slm_w21 + (size_t)(col - 1000) * 128, 128) + A.slm_tadd[n * 3 + 0];
        else if (col < 10000) v = dotL(sh_lhp2, A.slm_w22 + (size_t)(col - 3000) * 32, 32)   + A.slm_tadd[n * 3 + 1];
        else                  v = dot8(sh_lhp3, A.slm_w23 + (size_t)(col - 10000) * 8)        + A.slm_tadd[n * 3 + 2];
        acc += val * v;
    }

    A.out[(size_t)n * S_TOT + s] = y + 0.1f * acc;
}

// ---------------- host side ----------------
static inline char* bump(char*& p, size_t bytes) {
    char* r = p;
    p += (bytes + 255) & ~(size_t)255;
    return r;
}

static void launch_gemm(const _Float16* Af, const _Float16* W, const float* bias,
                        float* C, _Float16* Ch, int M, int N, int K, int relu,
                        hipStream_t stream) {
    int tilesN = (N + 15) / 16;
    int tiles = (M / 16) * tilesN;
    int blocks = (tiles + 7) / 8;   // 8 waves (wave32) per 256-thread block
    wmma_gemm_kernel<<<blocks, 256, 0, stream>>>(Af, W, bias, C, Ch, M, N, K, relu);
}

static void launch_f2h(const float* s, _Float16* d, int n, hipStream_t stream) {
    f2h_kernel<<<(n + 255) / 256, 256, 0, stream>>>(s, d, n);
}

extern "C" void kernel_launch(void* const* d_in, const int* in_sizes, int n_in,
                              void* d_out, int out_size, void* d_ws, size_t ws_size,
                              hipStream_t stream) {
    const float* x         = (const float*)d_in[0];
    const float* bn_gamma  = (const float*)d_in[1];
    const float* bn_beta   = (const float*)d_in[2];
    const float* dense_w   = (const float*)d_in[3];
    const float* dense_b   = (const float*)d_in[4];
    const float* h1_w      = (const float*)d_in[5];
    const float* h1_b      = (const float*)d_in[6];
    const float* h2_w      = (const float*)d_in[7];
    const float* h2_b      = (const float*)d_in[8];
    const float* wsd_head_w = (const float*)d_in[9];
    const float* wsd_t1_w1 = (const float*)d_in[10];
    const float* wsd_t1_w2 = (const float*)d_in[11];
    const float* wsd_t2_w1 = (const float*)d_in[12];
    const float* wsd_t2_w2 = (const float*)d_in[13];
    const float* wsd_t3_w1 = (const float*)d_in[14];
    const float* wsd_t3_w2 = (const float*)d_in[15];
    const float* slm_head_w = (const float*)d_in[16];
    const float* slm_t1_w1 = (const float*)d_in[17];
    const float* slm_t1_w2 = (const float*)d_in[18];
    const float* slm_t2_w1 = (const float*)d_in[19];
    const float* slm_t2_w2 = (const float*)d_in[20];
    const float* slm_t3_w1 = (const float*)d_in[21];
    const float* slm_t3_w2 = (const float*)d_in[22];
    const float* sv_vals   = (const float*)d_in[23];
    const int*   sv_cols   = (const int*)d_in[24];
    float* out = (float*)d_out;

    char* p = (char*)d_ws;
    float*     mu        = (float*)bump(p, D_IN * 4);
    float*     rsig      = (float*)bump(p, D_IN * 4);
    _Float16*  xh        = (_Float16*)bump(p, (size_t)N_ROWS * D_IN * 2);
    _Float16*  dense_wh  = (_Float16*)bump(p, (size_t)HID * D_IN * 2);
    _Float16*  h1_wh     = (_Float16*)bump(p, (size_t)HID * HID * 2);
    _Float16*  h2_wh     = (_Float16*)bump(p, (size_t)HID * HID * 2);
    _Float16*  wsd_headh = (_Float16*)bump(p, (size_t)HEADN * HID * 2);
    _Float16*  slm_headh = (_Float16*)bump(p, (size_t)HEADN * HID * 2);
    _Float16*  wsd_t1h   = (_Float16*)bump(p, (size_t)128 * HID * 2);
    _Float16*  wsd_t2h   = (_Float16*)bump(p, (size_t)32 * HID * 2);
    _Float16*  wsd_t3h   = (_Float16*)bump(p, (size_t)8 * HID * 2);
    _Float16*  slm_t1h   = (_Float16*)bump(p, (size_t)128 * HID * 2);
    _Float16*  slm_t2h   = (_Float16*)bump(p, (size_t)32 * HID * 2);
    _Float16*  slm_t3h   = (_Float16*)bump(p, (size_t)8 * HID * 2);
    _Float16*  y1h       = (_Float16*)bump(p, (size_t)N_ROWS * HID * 2);
    _Float16*  y2h       = (_Float16*)bump(p, (size_t)N_ROWS * HID * 2);
    _Float16*  hh        = (_Float16*)bump(p, (size_t)N_ROWS * HID * 2);
    float*     wsd_headlog = (float*)bump(p, (size_t)N_ROWS * HEADN * 4);
    float*     slm_headlog = (float*)bump(p, (size_t)N_ROWS * HEADN * 4);
    float*     wsd_hp1   = (float*)bump(p, (size_t)N_ROWS * 128 * 4);
    float*     wsd_hp2   = (float*)bump(p, (size_t)N_ROWS * 32 * 4);
    float*     wsd_hp3   = (float*)bump(p, (size_t)N_ROWS * 8 * 4);
    float*     slm_hp1   = (float*)bump(p, (size_t)N_ROWS * 128 * 4);
    float*     slm_hp2   = (float*)bump(p, (size_t)N_ROWS * 32 * 4);
    float*     slm_hp3   = (float*)bump(p, (size_t)N_ROWS * 8 * 4);
    float*     wsd_lseh  = (float*)bump(p, N_ROWS * 4);
    float*     slm_lseh  = (float*)bump(p, N_ROWS * 4);
    float*     wsd_tadd  = (float*)bump(p, N_ROWS * 3 * 4);
    float*     slm_tadd  = (float*)bump(p, N_ROWS * 3 * 4);
    (void)ws_size; (void)in_sizes; (void)n_in; (void)out_size;

    // 1) batchnorm
    bn_stats_kernel<<<(D_IN + 255) / 256, 256, 0, stream>>>(x, mu, rsig);
    bn_apply_kernel<<<(N_ROWS * D_IN + 255) / 256, 256, 0, stream>>>(x, mu, rsig, bn_gamma, bn_beta, xh);

    // 2) convert weights to f16 for WMMA
    launch_f2h(dense_w, dense_wh, HID * D_IN, stream);
    launch_f2h(h1_w, h1_wh, HID * HID, stream);
    launch_f2h(h2_w, h2_wh, HID * HID, stream);
    launch_f2h(wsd_head_w, wsd_headh, HEADN * HID, stream);
    launch_f2h(slm_head_w, slm_headh, HEADN * HID, stream);
    launch_f2h(wsd_t1_w1, wsd_t1h, 128 * HID, stream);
    launch_f2h(wsd_t2_w1, wsd_t2h, 32 * HID, stream);
    launch_f2h(wsd_t3_w1, wsd_t3h, 8 * HID, stream);
    launch_f2h(slm_t1_w1, slm_t1h, 128 * HID, stream);
    launch_f2h(slm_t2_w1, slm_t2h, 32 * HID, stream);
    launch_f2h(slm_t3_w1, slm_t3h, 8 * HID, stream);

    // 3) WMMA GEMM chain
    launch_gemm(xh,  dense_wh, dense_b, nullptr, y1h, N_ROWS, HID, D_IN, 0, stream);
    launch_gemm(y1h, h1_wh,    h1_b,    nullptr, y2h, N_ROWS, HID, HID, 1, stream);
    launch_gemm(y2h, h2_wh,    h2_b,    nullptr, hh,  N_ROWS, HID, HID, 0, stream);

    launch_gemm(hh, wsd_headh, nullptr, wsd_headlog, nullptr, N_ROWS, HEADN, HID, 0, stream);
    launch_gemm(hh, slm_headh, nullptr, slm_headlog, nullptr, N_ROWS, HEADN, HID, 0, stream);
    launch_gemm(hh, wsd_t1h, nullptr, wsd_hp1, nullptr, N_ROWS, 128, HID, 0, stream);
    launch_gemm(hh, wsd_t2h, nullptr, wsd_hp2, nullptr, N_ROWS, 32,  HID, 0, stream);
    launch_gemm(hh, wsd_t3h, nullptr, wsd_hp3, nullptr, N_ROWS, 8,   HID, 0, stream);
    launch_gemm(hh, slm_t1h, nullptr, slm_hp1, nullptr, N_ROWS, 128, HID, 0, stream);
    launch_gemm(hh, slm_t2h, nullptr, slm_hp2, nullptr, N_ROWS, 32,  HID, 0, stream);
    launch_gemm(hh, slm_t3h, nullptr, slm_hp3, nullptr, N_ROWS, 8,   HID, 0, stream);

    // 4) log-softmax normalizers
    row_lse_kernel<<<N_ROWS, 256, 0, stream>>>(wsd_headlog, wsd_lseh);
    row_lse_kernel<<<N_ROWS, 256, 0, stream>>>(slm_headlog, slm_lseh);

    tail_lse_kernel<<<N_ROWS, 256, 0, stream>>>(wsd_hp1, 128, wsd_t1_w2, 2000,   wsd_headlog, wsd_lseh, 0, wsd_tadd);
    tail_lse_kernel<<<N_ROWS, 256, 0, stream>>>(wsd_hp2, 32,  wsd_t2_w2, 7000,   wsd_headlog, wsd_lseh, 1, wsd_tadd);
    tail_lse_kernel<<<N_ROWS, 256, 0, stream>>>(wsd_hp3, 8,   wsd_t3_w2, T3_WSD, wsd_headlog, wsd_lseh, 2, wsd_tadd);
    tail_lse_kernel<<<N_ROWS, 256, 0, stream>>>(slm_hp1, 128, slm_t1_w2, 2000,   slm_headlog, slm_lseh, 0, slm_tadd);
    tail_lse_kernel<<<N_ROWS, 256, 0, stream>>>(slm_hp2, 32,  slm_t2_w2, 7000,   slm_headlog, slm_lseh, 1, slm_tadd);
    tail_lse_kernel<<<N_ROWS, 256, 0, stream>>>(slm_hp3, 8,   slm_t3_w2, T3_SLM, slm_headlog, slm_lseh, 2, slm_tadd);

    // 5) fused final output
    FinalArgs fa;
    fa.wsd_headlog = wsd_headlog; fa.wsd_lseh = wsd_lseh; fa.wsd_tadd = wsd_tadd;
    fa.wsd_hp1 = wsd_hp1; fa.wsd_hp2 = wsd_hp2; fa.wsd_hp3 = wsd_hp3;
    fa.wsd_w21 = wsd_t1_w2; fa.wsd_w22 = wsd_t2_w2; fa.wsd_w23 = wsd_t3_w2;
    fa.slm_headlog = slm_headlog; fa.slm_lseh = slm_lseh; fa.slm_tadd = slm_tadd;
    fa.slm_hp1 = slm_hp1; fa.slm_hp2 = slm_hp2; fa.slm_hp3 = slm_hp3;
    fa.slm_w21 = slm_t1_w2; fa.slm_w22 = slm_t2_w2; fa.slm_w23 = slm_t3_w2;
    fa.sv_vals = sv_vals; fa.sv_cols = sv_cols; fa.out = out;

    dim3 grid((S_TOT + 255) / 256, N_ROWS);
    final_out_kernel<<<grid, 256, 0, stream>>>(fa);
}
